// Decouple_44822278701619
// MI455X (gfx1250) — matrix operand order repaired and checked by reading, not compiled, not run on hardware
//
#include <hip/hip_runtime.h>
#include <hip/hip_bf16.h>

typedef __attribute__((ext_vector_type(16))) _Float16 v16h;
typedef __attribute__((ext_vector_type(8)))  float    v8f;

union PackU { unsigned int u; _Float16 h[2]; };
union FragU { v16h v; uint4 q[2]; };

static __device__ __forceinline__ v8f wmma_f16(v16h a, v16h b, v8f c) {
  // D = A(16x32 f16) * B(32x16 f16) + C(16x16 f32)
  return __builtin_amdgcn_wmma_f32_16x16x32_f16(false, a, false, b, (short)0, c, false, false);
}

// ---------------------------------------------------------------------------
// k0: prepare WMMA A-fragments (hi/lo f16 split) for conv_o (folded) & conv_r,
//     plus per-channel sum of conv_m_w.
// A-fragment per-lane layout (16x32 f16): lane<16 row M=lane, K pairs
//   v0..3 -> K{0..7}, v4..7 -> K{16..23}; lane>=16 row M=lane-16, K +8.
// ---------------------------------------------------------------------------
__global__ void k0_prep(const float* __restrict__ wo, const float* __restrict__ wm,
                        const float* __restrict__ wr,
                        _Float16* __restrict__ afragO, _Float16* __restrict__ afragR,
                        float* __restrict__ wsum)
{
  const int tid = threadIdx.x;
  // conv_o fragments: [mtile5][tap9][chunk2][plane2][lane32][16 halves]
  for (int idx = tid; idx < 5*9*2*2*32*16; idx += blockDim.x) {
    int hh = idx & 15;  int t1 = idx >> 4;
    int lane = t1 & 31; int t2 = t1 >> 5;
    int plane = t2 & 1; int t3 = t2 >> 1;
    int ch = t3 & 1;    int t4 = t3 >> 1;
    int tp = t4 % 9;    int m  = t4 / 9;
    int v = hh >> 1, pel = hh & 1;
    int kk = ((v & 3) << 1) + ((v >> 2) << 4) + ((lane >> 4) << 3) + pel;
    int c = ch*32 + kk;
    int o = m*16 + (lane & 15);
    float val = 0.f;
    if (o < 72) {
      const float* wp = wo + (o*64 + c)*9;
      val = wp[tp];
      if (tp == 4) {           // fold the 1x1 (kernel-summed) conv into center tap
        float s = 0.f;
        for (int q = 0; q < 9; ++q) s += wp[q];
        val -= s;
      }
    }
    _Float16 hi = (_Float16)val;
    afragO[idx] = (plane == 0) ? hi : (_Float16)(val - (float)hi);
  }
  // conv_r fragments: [mtile4][chunk5][plane2][lane32][16 halves], K padded 136->160
  for (int idx = tid; idx < 4*5*2*32*16; idx += blockDim.x) {
    int hh = idx & 15;  int t1 = idx >> 4;
    int lane = t1 & 31; int t2 = t1 >> 5;
    int plane = t2 & 1; int t3 = t2 >> 1;
    int ch = t3 % 5;    int m  = t3 / 5;
    int v = hh >> 1, pel = hh & 1;
    int kk = ((v & 3) << 1) + ((v >> 2) << 4) + ((lane >> 4) << 3) + pel;
    int k = ch*32 + kk;
    int o = m*16 + (lane & 15);
    float val = (k < 136) ? wr[o*136 + k] : 0.f;
    _Float16 hi = (_Float16)val;
    afragR[idx] = (plane == 0) ? hi : (_Float16)(val - (float)hi);
  }
  if (tid < 64) {
    float s = 0.f;
    for (int q = 0; q < 9; ++q) s += wm[tid*9 + q];
    wsum[tid] = s;
  }
}

// ---------------------------------------------------------------------------
// kT: NCHW -> NHWC transpose of x + per-pixel channel sums (mask input)
// ---------------------------------------------------------------------------
__global__ void kT_nchw2nhwc(const float* __restrict__ x, float* __restrict__ xt,
                             float* __restrict__ sums)
{
  __shared__ float tile[64][65];
  const int b = blockIdx.z, h = blockIdx.y, w0 = blockIdx.x*64;
  const int tid = threadIdx.x;
  const int ci = tid >> 6, px = tid & 63;
  for (int cc = 0; cc < 16; ++cc) {
    int c = ci*16 + cc;
    tile[c][px] = x[(((size_t)(b*64 + c)*256 + h)*256) + w0 + px];
  }
  __syncthreads();
  if (tid < 64) {
    float s = 0.f;
    for (int c = 0; c < 64; ++c) s += tile[c][tid];
    sums[((size_t)(b*256 + h)*256) + w0 + tid] = s;
  }
  for (int pass = 0; pass < 16; ++pass) {
    int p = (tid >> 6) + pass*4;
    int c = tid & 63;
    xt[(((size_t)(b*256 + h)*256 + w0 + p)*64) + c] = tile[c][p];
  }
}

// ---------------------------------------------------------------------------
// k2: t = conv3x3(x, folded W_o) + bias, via implicit GEMM (WMMA f16 hi/lo).
// Block = 128 thr (4 waves), tile = 64 output pixels of one row, all 72 ch.
// LDS layout: [row3][col66][plane2][cpair32], col stride padded to 68 dwords
// so each lane's 8 B-fragment dwords sit in two 4-dword runs 8 apart -> two ds_load_b128 per
// plane, no VALU repacking. (16B-aligned: all index terms are multiples of 4.)
// B fragment (32x16): lane<16 col N=lane, v0..3 -> K{0..7}, v4..7 -> K{16..23}; lane>=16 K+8.
// ---------------------------------------------------------------------------
#define K2_COLS 66
#define K2_CSTRIDE 68   // 64 dwords (2 planes x 32 cpairs) + 4 pad
__global__ void k2_conv_o(const float* __restrict__ x, const float* __restrict__ bias,
                          const _Float16* __restrict__ afragO,
                          float* __restrict__ t_nhwc)
{
  extern __shared__ unsigned int smem[];          // 3*66*68 dwords = 53856 B
  __shared__ float s_bias[80];
  const int b = blockIdx.z, h = blockIdx.y, w0 = blockIdx.x*64;
  const int tid = threadIdx.x;
  if (tid < 80) s_bias[tid] = (tid < 72) ? bias[tid] : 0.f;

  for (int idx = tid; idx < 32*3*K2_COLS; idx += blockDim.x) {
    int cp  = idx / (3*K2_COLS);
    int rem = idx % (3*K2_COLS);
    int r = rem / K2_COLS, col = rem % K2_COLS;
    int hr = h - 1 + r;
    int wc = w0 - 1 + col;
    float v0 = 0.f, v1 = 0.f;
    if (hr >= 0 && hr < 256 && wc >= 0 && wc < 256) {
      const float* xp = x + (((size_t)(b*64 + cp*2)*256 + hr)*256) + wc;
      v0 = xp[0];
      v1 = xp[65536];                              // next channel plane
    }
    PackU phi, plo;
    _Float16 h0 = (_Float16)v0, h1 = (_Float16)v1;
    phi.h[0] = h0; phi.h[1] = h1;
    plo.h[0] = (_Float16)(v0 - (float)h0);
    plo.h[1] = (_Float16)(v1 - (float)h1);
    int base = (r*K2_COLS + col)*K2_CSTRIDE;
    smem[base + cp]      = phi.u;                  // plane 0 (hi)
    smem[base + 32 + cp] = plo.u;                  // plane 1 (lo)
  }
  __syncthreads();

  const int wave = tid >> 5, lane = tid & 31;
  const int nlo = lane & 15, half = lane >> 4;
  const int lc = wave*16 + nlo;                    // local pixel 0..63
  v8f acc[5];
  for (int m = 0; m < 5; ++m) for (int e = 0; e < 8; ++e) acc[m][e] = 0.f;

  for (int tp = 0; tp < 9; ++tp) {
    const int i = tp/3, j = tp%3;
    for (int ch = 0; ch < 2; ++ch) {
      FragU bhi, blo;
      const int base = (i*K2_COLS + lc + j)*K2_CSTRIDE + ch*16 + half*4;
      bhi.q[0] = *(const uint4*)&smem[base];
      bhi.q[1] = *(const uint4*)&smem[base + 8];
      blo.q[0] = *(const uint4*)&smem[base + 32];
      blo.q[1] = *(const uint4*)&smem[base + 40];
      for (int m = 0; m < 5; ++m) {
        const _Float16* ap = afragO + (size_t)((((m*9 + tp)*2 + ch)*2 + 0)*32 + lane)*16;
        v16h ahi = *(const v16h*)ap;
        v16h alo = *(const v16h*)(ap + 512);       // plane stride = 32 lanes * 16 halves
        acc[m] = wmma_f16(ahi, bhi.v, acc[m]);     // hi*hi
        acc[m] = wmma_f16(ahi, blo.v, acc[m]);     // hi*lo
        acc[m] = wmma_f16(alo, bhi.v, acc[m]);     // lo*hi  (~fp32 accuracy)
      }
    }
  }
  // D layout: VGPR r -> row r (lanes<16) / r+8 (lanes>=16), col = lane%16
  const int wpix = w0 + lc;
  float* top = t_nhwc + ((size_t)((b*256 + h)*256 + wpix))*72;
  for (int m = 0; m < 5; ++m)
    for (int r = 0; r < 8; ++r) {
      int M = m*16 + r + half*8;
      if (M < 72) top[M] = acc[m][r] + s_bias[M];
    }
}

// ---------------------------------------------------------------------------
// k3: deform gather (NHWC, contiguous 16-ch groups) -> cat[xo, masked t] in
//     LDS (f16 hi/lo pair-packed) -> WMMA conv_r (136->64) -> tanh -> NCHW out
// Block = 128 thr (4 waves), tile = 64 pixels of one row.
// LDS layout: [px64][plane2][kp80], px stride padded to 164 dwords -> each
// lane's 8 fragment dwords in two 4-dword runs 8 apart (two ds_load_b128 per plane); writes of
// the cat vector are contiguous runs the compiler can merge.
// ---------------------------------------------------------------------------
#define K3_PSTRIDE 164  // 160 dwords (2 planes x 80 kpairs) + 4 pad
__global__ void k3_fused(const float* __restrict__ xt, const float* __restrict__ t_nhwc,
                         const float* __restrict__ sums, const float* __restrict__ wm,
                         const float* __restrict__ wsum, const _Float16* __restrict__ afragR,
                         float* __restrict__ out)
{
  extern __shared__ unsigned int cat[];            // 64*164 dwords = 41984 B
  const int b = blockIdx.z, h = blockIdx.y, w0 = blockIdx.x*64;
  const int tid = threadIdx.x;

  for (int u = tid; u < 256; u += blockDim.x) {    // unit = (pixel, deform-group)
    const int px = u & 63, g = u >> 6;
    const int wpix = w0 + px;
    const size_t pixbase = (size_t)((b*256 + h)*256 + wpix);
    const float* tp = t_nhwc + pixbase*72;
    const float* xp = xt + pixbase*64 + g*16;
    unsigned int* catp = cat + px*K3_PSTRIDE;
    float accv[16];
    for (int c = 0; c < 16; ++c) accv[c] = 0.f;

    for (int m = 0; m < 9; ++m) {
      const int i = m/3, j = m%3;
      float dy = tp[(g*9 + m)*2 + 0];
      float dx = tp[(g*9 + m)*2 + 1];
      float py  = (float)(h - 1 + i) + dy;
      float pxf = (float)(wpix - 1 + j) + dx;
      float y0f = floorf(py), x0f = floorf(pxf);
      float wy1 = py - y0f, wy0 = 1.f - wy1;
      float wx1 = pxf - x0f, wx0 = 1.f - wx1;
      int y0 = (int)y0f, x0 = (int)x0f;
      float samp[16];
      for (int c = 0; c < 16; ++c) samp[c] = 0.f;
      for (int cy = 0; cy < 2; ++cy) {
        int yy = y0 + cy;
        if (yy < 0 || yy > 255) continue;
        float wy = cy ? wy1 : wy0;
        for (int cx = 0; cx < 2; ++cx) {
          int xx = x0 + cx;
          if (xx < 0 || xx > 255) continue;
          float wgt = wy * (cx ? wx1 : wx0);
          const float4* src = (const float4*)(xt + (((size_t)(b*256 + yy)*256 + xx)*64) + g*16);
          for (int q = 0; q < 4; ++q) {
            float4 f = src[q];
            samp[q*4+0] += wgt*f.x; samp[q*4+1] += wgt*f.y;
            samp[q*4+2] += wgt*f.z; samp[q*4+3] += wgt*f.w;
          }
        }
      }
      const float* wmp = wm + (g*16)*9 + m;
      for (int c = 0; c < 16; ++c) accv[c] += wmp[c*9] * samp[c];
    }
    // xo = deform - x * sum(w_m); cat K rows 0..63 (hi/lo split), contiguous
    for (int cp = 0; cp < 8; ++cp) {
      float a0 = accv[cp*2]   - xp[cp*2]   * wsum[g*16 + cp*2];
      float a1 = accv[cp*2+1] - xp[cp*2+1] * wsum[g*16 + cp*2+1];
      PackU phi, plo;
      _Float16 h0 = (_Float16)a0, h1 = (_Float16)a1;
      phi.h[0] = h0; phi.h[1] = h1;
      plo.h[0] = (_Float16)(a0 - (float)h0);
      plo.h[1] = (_Float16)(a1 - (float)h1);
      int kp = g*8 + cp;
      catp[kp]      = phi.u;
      catp[80 + kp] = plo.u;
    }
    // masked t: cat K rows 64..135 (this unit handles 18 of the 72 t channels)
    float maskf = (sums[pixbase] > 0.f) ? 1.f : 0.f;
    for (int pj = 0; pj < 9; ++pj) {
      int j0 = g*18 + pj*2;
      float a0 = tp[j0] * maskf, a1 = tp[j0+1] * maskf;
      PackU phi, plo;
      _Float16 h0 = (_Float16)a0, h1 = (_Float16)a1;
      phi.h[0] = h0; phi.h[1] = h1;
      plo.h[0] = (_Float16)(a0 - (float)h0);
      plo.h[1] = (_Float16)(a1 - (float)h1);
      int kp = 32 + g*9 + pj;
      catp[kp]      = phi.u;
      catp[80 + kp] = plo.u;
    }
    for (int pz = 0; pz < 3; ++pz) {               // zero-pad K 136..159
      int kp = 68 + g*3 + pz;
      catp[kp]      = 0u;
      catp[80 + kp] = 0u;
    }
  }
  __syncthreads();

  const int wave = tid >> 5, lane = tid & 31;
  const int nlo = lane & 15, half = lane >> 4;
  const int lc = wave*16 + nlo;
  v8f acc[4];
  for (int m = 0; m < 4; ++m) for (int e = 0; e < 8; ++e) acc[m][e] = 0.f;

  for (int ch = 0; ch < 5; ++ch) {
    FragU bhi, blo;
    const int base = lc*K3_PSTRIDE + ch*16 + half*4;
    bhi.q[0] = *(const uint4*)&cat[base];
    bhi.q[1] = *(const uint4*)&cat[base + 8];
    blo.q[0] = *(const uint4*)&cat[base + 80];
    blo.q[1] = *(const uint4*)&cat[base + 88];
    for (int m = 0; m < 4; ++m) {
      const _Float16* ap = afragR + (size_t)(((m*5 + ch)*2 + 0)*32 + lane)*16;
      v16h ahi = *(const v16h*)ap;
      v16h alo = *(const v16h*)(ap + 512);
      acc[m] = wmma_f16(ahi, bhi.v, acc[m]);
      acc[m] = wmma_f16(ahi, blo.v, acc[m]);
      acc[m] = wmma_f16(alo, bhi.v, acc[m]);
    }
  }
  const int wpix = w0 + lc;
  for (int m = 0; m < 4; ++m)
    for (int r = 0; r < 8; ++r) {
      int M = m*16 + r + half*8;
      out[(((size_t)(b*64 + M)*256 + h)*256) + wpix] = tanhf(acc[m][r]);
    }
}

// ---------------------------------------------------------------------------
extern "C" void kernel_launch(void* const* d_in, const int* in_sizes, int n_in,
                              void* d_out, int out_size, void* d_ws, size_t ws_size,
                              hipStream_t stream)
{
  const float* x  = (const float*)d_in[0];   // [4,64,256,256]
  const float* wo = (const float*)d_in[1];   // [72,64,3,3]
  const float* bo = (const float*)d_in[2];   // [72]
  const float* wm = (const float*)d_in[3];   // [64,1,3,3]
  const float* wr = (const float*)d_in[4];   // [64,136]
  float* out = (float*)d_out;                // [4,64,256,256]

  char* ws = (char*)d_ws;
  size_t off = 0;
  auto take = [&](size_t bytes) -> char* {
    char* p = ws + off;
    off = (off + bytes + 255) & ~(size_t)255;
    return p;
  };
  float*    t_nhwc = (float*)   take((size_t)4*256*256*72*4);   // 75.5 MB
  float*    x_nhwc = (float*)   take((size_t)4*256*256*64*4);   // 67.1 MB
  float*    sums   = (float*)   take((size_t)4*256*256*4);      //  1.0 MB
  _Float16* afragO = (_Float16*)take((size_t)5*9*2*2*32*16*2);  // 184 KB
  _Float16* afragR = (_Float16*)take((size_t)4*5*2*32*16*2);    //  40 KB
  float*    wsum   = (float*)   take((size_t)64*4);
  if (off > ws_size) return;

  k0_prep     <<<dim3(1),       dim3(256), 0,     stream>>>(wo, wm, wr, afragO, afragR, wsum);
  kT_nchw2nhwc<<<dim3(4,256,4), dim3(256), 0,     stream>>>(x, x_nhwc, sums);
  k2_conv_o   <<<dim3(4,256,4), dim3(128), 53856, stream>>>(x, bo, afragO, t_nhwc);
  k3_fused    <<<dim3(4,256,4), dim3(128), 41984, stream>>>(x_nhwc, t_nhwc, sums, wm, wsum,
                                                            afragR, out);
}
